// Multihead_Attention_1657857376658
// MI455X (gfx1250) — matrix operand order repaired and checked by reading, not compiled, not run on hardware
//
#include <hip/hip_runtime.h>

// ---------------------------------------------------------------------------
// Multihead attention layer (LN -> QKV proj -> causal flash attn -> out proj
// + residual) for MI455X / gfx1250.
// Matrix ops: v_wmma_f32_16x16x32_bf16.  W tiles staged to LDS with the
// CDNA5 async-to-LDS engine (ASYNCcnt) and double-buffered.
// B=2, S=2048, E=1024, H=16, D=64.
// ---------------------------------------------------------------------------

static constexpr int kB = 2;
static constexpr int kS = 2048;
static constexpr int kE = 1024;
static constexpr int kH = 16;
static constexpr int kD = 64;
static constexpr int kM = kB * kS;           // 4096 rows in the GEMMs

typedef __bf16        v16bf __attribute__((ext_vector_type(16)));
typedef float         v8f   __attribute__((ext_vector_type(8)));
typedef unsigned int  v4u   __attribute__((ext_vector_type(4)));
typedef int           v4i   __attribute__((ext_vector_type(4)));

#define GLOBAL_AS __attribute__((address_space(1)))
#define LDS_AS    __attribute__((address_space(3)))

union Frag16 {           // 16 bf16 = 32 bytes = one WMMA A/B operand (8 VGPRs)
    v16bf v;
    v4u   u[2];
};

__device__ inline v16bf load_frag(const unsigned short* p0,
                                  const unsigned short* p1) {
    Frag16 f;
    f.u[0] = *(const v4u*)p0;    // 128-bit load
    f.u[1] = *(const v4u*)p1;    // 128-bit load
    return f.v;
}

__device__ inline v8f wmma_bf16(v16bf a, v16bf b, v8f c) {
    // D(16x16,f32) = A(16x32,bf16) * B(32x16,bf16) + C
    return __builtin_amdgcn_wmma_f32_16x16x32_bf16(
        false, a, false, b, (short)0, c, false, false);
}

// fp32 -> bf16 bits, round-to-nearest-even (no reliance on __bf16 arithmetic)
__device__ inline unsigned short f2bf(float f) {
    union { float f; unsigned u; } a;
    a.f = f;
    unsigned u = a.u;
    unsigned r = (u + 0x7FFFu + ((u >> 16) & 1u)) >> 16;
    return (unsigned short)r;
}

// ---- CDNA5 async global->LDS copy (ASYNCcnt path), 16 bytes per lane ------
#if defined(__has_builtin)
#if __has_builtin(__builtin_amdgcn_global_load_async_to_lds_b128)
#define HAVE_ASYNC_B128 1
#endif
#endif

__device__ inline void async_copy_b128(const unsigned short* gsrc,
                                       unsigned short* ldst) {
#if defined(HAVE_ASYNC_B128)
    // signature: (int4 AS1* gaddr, int4 AS3* lds, imm offset, imm cpol)
    __builtin_amdgcn_global_load_async_to_lds_b128(
        (GLOBAL_AS v4i*)gsrc, (LDS_AS v4i*)ldst, 0, 0);
#else
    unsigned lds_off = (unsigned)(unsigned long long)(LDS_AS void*)ldst;
    asm volatile("global_load_async_to_lds_b128 %0, %1, off"
                 :: "v"(lds_off), "v"(gsrc) : "memory");
#endif
}

template <int N>
__device__ inline void wait_asynccnt() {
#if defined(__has_builtin)
#if __has_builtin(__builtin_amdgcn_s_wait_asynccnt)
    __builtin_amdgcn_s_wait_asynccnt(N);
#define DID_WAIT_ASYNC 1
#endif
#endif
#if !defined(DID_WAIT_ASYNC)
    asm volatile("s_wait_asynccnt %0" :: "i"(N) : "memory");
#endif
#undef DID_WAIT_ASYNC
}

// ---------------------------------------------------------------------------
// Kernel 1: fp32 -> bf16 weight conversion
// ---------------------------------------------------------------------------
__global__ __launch_bounds__(256)
void cvt_bf16_kernel(const float* __restrict__ src,
                     unsigned short* __restrict__ dst, int n) {
    int i = blockIdx.x * 256 + threadIdx.x;
    if (i < n) dst[i] = f2bf(src[i]);
}

// ---------------------------------------------------------------------------
// Kernel 2: LayerNorm of one E=1024 row per 256-thread block.
// ---------------------------------------------------------------------------
__global__ __launch_bounds__(256)
void ln_kernel(const float* __restrict__ x,
               const float* __restrict__ gamma,
               const float* __restrict__ beta,
               unsigned short* __restrict__ ybf,
               float* __restrict__ yf32) {
    __shared__ float red[16];
    const int row = blockIdx.x;
    const int t   = threadIdx.x;
    const float* xr = x + (size_t)row * kE;

    float v[4];
    float s = 0.f, s2 = 0.f;
#pragma unroll
    for (int i = 0; i < 4; ++i) {
        float a = xr[t + i * 256];
        v[i] = a; s += a; s2 += a * a;
    }
#pragma unroll
    for (int d = 16; d >= 1; d >>= 1) {
        s  += __shfl_xor(s,  d, 32);
        s2 += __shfl_xor(s2, d, 32);
    }
    if ((t & 31) == 0) { red[t >> 5] = s; red[8 + (t >> 5)] = s2; }
    __syncthreads();
    float fs = 0.f, fs2 = 0.f;
#pragma unroll
    for (int i = 0; i < 8; ++i) { fs += red[i]; fs2 += red[8 + i]; }

    const float mu   = fs  * (1.0f / kE);
    const float var  = fs2 * (1.0f / kE) - mu * mu;
    const float rstd = rsqrtf(var + 1e-5f);

#pragma unroll
    for (int i = 0; i < 4; ++i) {
        int idx = t + i * 256;
        float yn = (v[i] - mu) * rstd * gamma[idx] + beta[idx];
        ybf[(size_t)row * kE + idx] = f2bf(yn);
        if (yf32) yf32[(size_t)row * kE + idx] = yn;
    }
}

// ---------------------------------------------------------------------------
// GEMM core: y[m,n] = sum_k A[m,k]*W[n,k]   (M x 1024 x 1024)
// Block = 256 thr (8 waves) -> 128(M) x 64(N) tile.
// W k-slices (64x32 bf16 = 4KB) staged to LDS via async-to-LDS, double
// buffered (ASYNCcnt), shared by all 8 waves.  A fragment register
// double-buffered.  4 WMMA accumulators per wave (A-fragment reuse x4).
// ---------------------------------------------------------------------------
struct GemmCtx {
    v8f acc[4];
    int lo, hi, m0, n0, koffA, ksegB;
};

__device__ inline void gemm_core(const unsigned short* __restrict__ A,
                                 const unsigned short* __restrict__ W,
                                 unsigned short (*wtile)[64 * 32],
                                 GemmCtx& c) {
    const int t    = threadIdx.x;
    const int lane = t & 31;
    const int wave = t >> 5;
    c.lo = lane & 15;
    c.hi = lane >> 4;
    c.m0 = blockIdx.y * 128 + wave * 16;
    c.n0 = blockIdx.x * 64;
    c.koffA = c.hi * 16;     // A-frag: 16 contiguous K starting at koffA
    c.ksegB = c.hi * 16;     // B-frag: 16 contiguous K starting at ksegB

    // cooperative W-tile copy mapping: 64 rows x 4 chunks of 16B
    const int cn = t >> 2;
    const int cc = t & 3;
    const unsigned short* wsrc = W + (size_t)(c.n0 + cn) * kE + cc * 8;
    unsigned short*       wdst = &wtile[0][0] + cn * 32 + cc * 8;

    const unsigned short* arow = A + (size_t)(c.m0 + c.lo) * kE;

    c.acc[0] = v8f{}; c.acc[1] = v8f{}; c.acc[2] = v8f{}; c.acc[3] = v8f{};

    constexpr int NIT = kE / 32;                   // 32 k-steps
    async_copy_b128(wsrc, wdst);                   // fill buffer 0
    v16bf a_cur = load_frag(arow + c.koffA, arow + 8 + c.koffA);

    for (int it = 0; it < NIT; ++it) {
        const int cur = it & 1;
        if (it + 1 < NIT) {
            // kick off next tile fill, then wait only for the older fill
            async_copy_b128(wsrc + (it + 1) * 32,
                            &wtile[1 - cur][0] + cn * 32 + cc * 8);
            wait_asynccnt<1>();
        } else {
            wait_asynccnt<0>();
        }
        __syncthreads();                           // tile[cur] visible to all

        const v16bf a_use = a_cur;
        if (it + 1 < NIT) {                        // prefetch next A fragment
            const unsigned short* ap = arow + (it + 1) * 32;
            a_cur = load_frag(ap + c.koffA, ap + 8 + c.koffA);
        }

        const unsigned short* wl = &wtile[cur][0] + c.lo * 32 + c.ksegB;
#pragma unroll
        for (int nt = 0; nt < 4; ++nt) {
            const unsigned short* wp = wl + nt * 16 * 32;
            c.acc[nt] = wmma_bf16(a_use, load_frag(wp, wp + 8), c.acc[nt]);
        }
        __syncthreads();                           // done reading tile[cur]
    }
}

// ---------------------------------------------------------------------------
// Kernel 3: QKV projection (+bias), output split to heads:
//   vtrans=0: [B,H,S,D] (Q,K)    vtrans=1: [B,H,D,S] (V transposed)
// ---------------------------------------------------------------------------
__global__ __launch_bounds__(256)
void proj_gemm_kernel(const unsigned short* __restrict__ A,
                      const unsigned short* __restrict__ W,
                      const float* __restrict__ bias,
                      unsigned short* __restrict__ out, int vtrans) {
    __shared__ __align__(16) unsigned short wtile[2][64 * 32];  // 2 x 4KB
    GemmCtx c;
    gemm_core(A, W, wtile, c);

    // Epilogue: C layout -> lane holds column N=lo, rows M = r + hi*8
#pragma unroll
    for (int nt = 0; nt < 4; ++nt) {
        const int ncol = c.n0 + nt * 16 + c.lo;
        const float bv = bias[ncol];
        const int hh = ncol >> 6;        // /D
        const int dd = ncol & (kD - 1);
#pragma unroll
        for (int r = 0; r < 8; ++r) {
            const int row = c.m0 + r + c.hi * 8;
            const int bb  = row >> 11;               // /S
            const int ss  = row & (kS - 1);
            const float val = c.acc[nt][r] + bv;
            size_t dst;
            if (!vtrans)
                dst = (((size_t)(bb * kH + hh) * kS + ss) * kD + dd);
            else
                dst = (((size_t)(bb * kH + hh) * kD + dd) * kS + ss);
            out[dst] = f2bf(val);
        }
    }
}

// ---------------------------------------------------------------------------
// Kernel 5: output GEMM, fused bias + residual(qn) in fp32.
// ---------------------------------------------------------------------------
__global__ __launch_bounds__(256)
void out_gemm_kernel(const unsigned short* __restrict__ A,
                     const unsigned short* __restrict__ W,
                     const float* __restrict__ bias,
                     const float* __restrict__ resid,
                     float* __restrict__ out) {
    __shared__ __align__(16) unsigned short wtile[2][64 * 32];
    GemmCtx c;
    gemm_core(A, W, wtile, c);

#pragma unroll
    for (int nt = 0; nt < 4; ++nt) {
        const int ncol = c.n0 + nt * 16 + c.lo;
        const float bv = bias[ncol];
#pragma unroll
        for (int r = 0; r < 8; ++r) {
            const int row = c.m0 + r + c.hi * 8;
            const size_t idx = (size_t)row * kE + ncol;
            out[idx] = c.acc[nt][r] + bv + resid[idx];
        }
    }
}

// ---------------------------------------------------------------------------
// Kernel 4: causal flash attention, one wave per 16-row q tile.
// Q,K: [B,H,S,D] bf16, Vt: [B,H,D,S] bf16, O: [B,S,E] bf16.
// 32 keys/step: 4 WMMAs QK^T; V fragments loaded eagerly so their latency
// hides under the softmax VALU work; P goes through a 1KB LDS tile (C->A
// layout); 4 WMMAs P*V.  Single-wave workgroups: __syncthreads ~ DS fence.
// ---------------------------------------------------------------------------
__global__ __launch_bounds__(32)
void flash_attn_kernel(const unsigned short* __restrict__ Q,
                       const unsigned short* __restrict__ K,
                       const unsigned short* __restrict__ Vt,
                       unsigned short* __restrict__ O) {
    __shared__ __align__(16) unsigned short Pl[16 * 32];

    const int lane = threadIdx.x;
    const int lo   = lane & 15;
    const int hi   = lane >> 4;
    const int koffA = hi * 16;
    const int ksegB = hi * 16;

    const int wid = blockIdx.x;
    const int qt  = wid & (kS / 16 - 1);    // q tile 0..127
    const int bh  = wid >> 7;               // (b*H + h) 0..31
    const int bb  = bh >> 4;
    const int hh  = bh & (kH - 1);

    const unsigned short* Qb = Q  + (size_t)bh * kS * kD;
    const unsigned short* Kb = K  + (size_t)bh * kS * kD;
    const unsigned short* Vb = Vt + (size_t)bh * kD * kS;

    const unsigned short* qr = Qb + (size_t)(qt * 16 + lo) * kD;
    const v16bf aq0 = load_frag(qr + koffA,      qr + 8 + koffA);
    const v16bf aq1 = load_frag(qr + 32 + koffA, qr + 40 + koffA);

    v8f o0 = {}, o1 = {}, o2 = {}, o3 = {};
    float mrow[8], lrow[8];
#pragma unroll
    for (int r = 0; r < 8; ++r) { mrow[r] = -1e30f; lrow[r] = 0.f; }

    const int nkt = (qt * 16 + 16 + 31) >> 5;
    for (int kt = 0; kt < nkt; ++kt) {
        const int kb = kt * 32;

        // ---- scores: S(16x32) = Q(16x64) @ K^T(64x32) --------------------
        v8f s0 = {}, s1 = {};
        {
            const unsigned short* kr0 = Kb + (size_t)(kb + lo)      * kD + ksegB;
            const unsigned short* kr1 = Kb + (size_t)(kb + 16 + lo) * kD + ksegB;
            s0 = wmma_bf16(aq0, load_frag(kr0,      kr0 + 8),  s0);
            s0 = wmma_bf16(aq1, load_frag(kr0 + 32, kr0 + 40), s0);
            s1 = wmma_bf16(aq0, load_frag(kr1,      kr1 + 8),  s1);
            s1 = wmma_bf16(aq1, load_frag(kr1 + 32, kr1 + 40), s1);
        }

        // ---- V fragments: independent of softmax -> issue loads now ------
        const unsigned short* vr = Vb + (size_t)lo * kS + kb + ksegB;
        const v16bf bv0 = load_frag(vr,                       vr + 8);
        const v16bf bv1 = load_frag(vr + (size_t)16 * kS,     vr + (size_t)16 * kS + 8);
        const v16bf bv2 = load_frag(vr + (size_t)32 * kS,     vr + (size_t)32 * kS + 8);
        const v16bf bv3 = load_frag(vr + (size_t)48 * kS,     vr + (size_t)48 * kS + 8);

        const int key0 = kb + lo;
        const int key1 = key0 + 16;

        // ---- causal mask + online softmax, P (bf16) -> LDS ---------------
#pragma unroll
        for (int r = 0; r < 8; ++r) {
            const int qrow = qt * 16 + r + hi * 8;
            float v0 = (key0 <= qrow) ? s0[r] : -1e30f;
            float v1 = (key1 <= qrow) ? s1[r] : -1e30f;

            float mx = fmaxf(v0, v1);
            mx = fmaxf(mx, __shfl_xor(mx, 1, 32));
            mx = fmaxf(mx, __shfl_xor(mx, 2, 32));
            mx = fmaxf(mx, __shfl_xor(mx, 4, 32));
            mx = fmaxf(mx, __shfl_xor(mx, 8, 32));

            const float mn = fmaxf(mrow[r], mx);
            const float sc = __expf(mrow[r] - mn);
            mrow[r] = mn;

            const float p0 = __expf(v0 - mn);
            const float p1 = __expf(v1 - mn);
            float ps = p0 + p1;
            ps += __shfl_xor(ps, 1, 32);
            ps += __shfl_xor(ps, 2, 32);
            ps += __shfl_xor(ps, 4, 32);
            ps += __shfl_xor(ps, 8, 32);
            lrow[r] = lrow[r] * sc + ps;

            o0[r] *= sc; o1[r] *= sc; o2[r] *= sc; o3[r] *= sc;

            const int prow = r + hi * 8;
            Pl[prow * 32 + lo]      = f2bf(p0);
            Pl[prow * 32 + 16 + lo] = f2bf(p1);
        }

        __syncthreads();   // single-wave WG: DS ordering fence
        const v16bf ap = load_frag(&Pl[lo * 32 + koffA],
                                   &Pl[lo * 32 + 8 + koffA]);
        __syncthreads();

        // ---- O(16x64) += P(16x32) @ V(32x64) -----------------------------
        o0 = wmma_bf16(ap, bv0, o0);
        o1 = wmma_bf16(ap, bv1, o1);
        o2 = wmma_bf16(ap, bv2, o2);
        o3 = wmma_bf16(ap, bv3, o3);
    }

    // ---- normalize and store O as [B,S,E] bf16 ------------------------------
#pragma unroll
    for (int r = 0; r < 8; ++r) {
        const float inv = 1.0f / lrow[r];
        const int qrow = qt * 16 + r + hi * 8;
        unsigned short* op = O + ((size_t)bb * kS + qrow) * kE + hh * kD;
        op[lo]      = f2bf(o0[r] * inv);
        op[16 + lo] = f2bf(o1[r] * inv);
        op[32 + lo] = f2bf(o2[r] * inv);
        op[48 + lo] = f2bf(o3[r] * inv);
    }
}

// ---------------------------------------------------------------------------
// Host-side orchestration (graph-capture safe: only kernel launches on stream)
// ---------------------------------------------------------------------------
extern "C" void kernel_launch(void* const* d_in, const int* in_sizes, int n_in,
                              void* d_out, int out_size, void* d_ws, size_t ws_size,
                              hipStream_t stream) {
    const float* q = (const float*)d_in[0];
    const float* k = (const float*)d_in[1];
    const float* v = (const float*)d_in[2];
    // d_in[3] = h_mask (unused; causal mask computed analytically)
    const float* Wq = (const float*)d_in[4];
    const float* bq = (const float*)d_in[5];
    const float* Wk = (const float*)d_in[6];
    const float* bk = (const float*)d_in[7];
    const float* Wv = (const float*)d_in[8];
    const float* bv = (const float*)d_in[9];
    const float* Wo = (const float*)d_in[10];
    const float* bo = (const float*)d_in[11];
    const float* gamma = (const float*)d_in[12];
    const float* beta  = (const float*)d_in[13];
    float* out = (float*)d_out;

    // ---- workspace carve-up (~80 MB, 256B aligned slices) ----
    char* ws = (char*)d_ws;
    size_t off = 0;
    auto take = [&](size_t bytes) -> char* {
        char* p = ws + off;
        off += (bytes + 255) & ~(size_t)255;
        return p;
    };
    const size_t actBytes = (size_t)kM * kE * sizeof(unsigned short); // 8 MB
    const size_t wBytes   = (size_t)kE * kE * sizeof(unsigned short); // 2 MB

    unsigned short* qn_bf = (unsigned short*)take(actBytes);
    unsigned short* kn_bf = (unsigned short*)take(actBytes);
    unsigned short* vn_bf = (unsigned short*)take(actBytes);
    float*          qn_f  = (float*)take((size_t)kM * kE * sizeof(float));
    unsigned short* Wq_bf = (unsigned short*)take(wBytes);
    unsigned short* Wk_bf = (unsigned short*)take(wBytes);
    unsigned short* Wv_bf = (unsigned short*)take(wBytes);
    unsigned short* Wo_bf = (unsigned short*)take(wBytes);
    unsigned short* Qh    = (unsigned short*)take(actBytes);  // [B,H,S,D]
    unsigned short* Kh    = (unsigned short*)take(actBytes);  // [B,H,S,D]
    unsigned short* Vth   = (unsigned short*)take(actBytes);  // [B,H,D,S]
    unsigned short* Obf   = (unsigned short*)take(actBytes);  // [B,S,E]

    // 1) weights -> bf16
    {
        const int n = kE * kE;
        dim3 g((n + 255) / 256), b(256);
        cvt_bf16_kernel<<<g, b, 0, stream>>>(Wq, Wq_bf, n);
        cvt_bf16_kernel<<<g, b, 0, stream>>>(Wk, Wk_bf, n);
        cvt_bf16_kernel<<<g, b, 0, stream>>>(Wv, Wv_bf, n);
        cvt_bf16_kernel<<<g, b, 0, stream>>>(Wo, Wo_bf, n);
    }

    // 2) layernorm q,k,v (q also keeps fp32 copy for residual)
    {
        dim3 g(kM), b(256);
        ln_kernel<<<g, b, 0, stream>>>(q, gamma, beta, qn_bf, qn_f);
        ln_kernel<<<g, b, 0, stream>>>(k, gamma, beta, kn_bf, (float*)nullptr);
        ln_kernel<<<g, b, 0, stream>>>(v, gamma, beta, vn_bf, (float*)nullptr);
    }

    // 3) Q/K/V projections (V stored transposed per head)
    {
        dim3 g(kE / 64, kM / 128), b(256);
        proj_gemm_kernel<<<g, b, 0, stream>>>(qn_bf, Wq_bf, bq, Qh, 0);
        proj_gemm_kernel<<<g, b, 0, stream>>>(kn_bf, Wk_bf, bk, Kh, 0);
        proj_gemm_kernel<<<g, b, 0, stream>>>(vn_bf, Wv_bf, bv, Vth, 1);
    }

    // 4) causal flash attention: one wave per 16-row q tile
    {
        dim3 g(kB * kH * (kS / 16)), b(32);   // 4096 single-wave blocks
        flash_attn_kernel<<<g, b, 0, stream>>>(Qh, Kh, Vth, Obf);
    }

    // 5) output projection + bias + residual(qn)
    {
        dim3 g(kE / 64, kM / 128), b(256);
        out_gemm_kernel<<<g, b, 0, stream>>>(Obf, Wo_bf, bo, qn_f, out);
    }
}
